// TransformerBlock_61357902791250
// MI455X (gfx1250) — matrix operand order repaired and checked by reading, not compiled, not run on hardware
//
#include <hip/hip_runtime.h>
#include <stdint.h>

#define S_LEN 2048
#define D_DIM 2048
#define H_NUM 16
#define HD_DIM 128
#define I_DIM 8192

typedef __attribute__((ext_vector_type(16))) __bf16 v16bf;
typedef __attribute__((ext_vector_type(8)))  float  v8f;

union FragU { uint32_t u[8]; uint4 q[2]; v16bf v; };

// ---- CDNA5 async global->LDS copy (ASYNCcnt-tracked, no VGPR staging) ----
__device__ inline void async_g2l_b128(void* lds, const void* g) {
  uint32_t l = (uint32_t)(uintptr_t)lds;   // generic LDS addr: low 32 bits = LDS offset
  asm volatile("global_load_async_to_lds_b128 %0, %1, off" :: "v"(l), "v"(g) : "memory");
}
#define WAIT_ASYNCCNT(n) asm volatile("s_wait_asynccnt %0" :: "n"(n) : "memory")

// A-fragment (16x32 bf16, M x K), row-major source with leading dim ld.
// Per ISA 7.12.2: lanes 0-15/16-31 hold M=0..15; VGPR0-3: K = half*8 + 0..7,
// VGPR4-7: K = 16 + half*8 + 0..7  -> two 16B loads per lane.
__device__ inline v16bf load_frag_a(const __bf16* base, int row0, int k0, int ld, int lane) {
  int r = lane & 15, hl = lane >> 4;
  const __bf16* p = base + (row0 + r) * ld + k0 + hl * 8;
  FragU f;
  f.q[0] = *(const uint4*)p;
  f.q[1] = *(const uint4*)(p + 16);
  return f.v;
}

// B-fragment (32x16 bf16, K x N), source stored as Bt row-major [N, K].
// Same K order as A: VGPR0-3: K = half*8 + 0..7, VGPR4-7: K = 16 + half*8 + 0..7.
__device__ inline v16bf load_frag_b(const __bf16* base, int col0, int k0, int ld, int lane) {
  int c = lane & 15, hl = lane >> 4;
  const __bf16* p = base + (col0 + c) * ld + k0 + hl * 8;
  FragU f;
  f.q[0] = *(const uint4*)p;
  f.q[1] = *(const uint4*)(p + 16);
  return f.v;
}

__device__ inline v8f wmma_bf16(v16bf a, v16bf b, v8f c) {
  return __builtin_amdgcn_wmma_f32_16x16x32_bf16(false, a, false, b, (short)0, c, false, false);
}

// ---------------- RMSNorm: fp32 [S,D] -> bf16 [S,D] ----------------
__global__ __launch_bounds__(256) void rmsnorm_kernel(
    const float* __restrict__ x, const float* __restrict__ w, __bf16* __restrict__ out) {
  int row = blockIdx.x;
  const float* xr = x + (size_t)row * D_DIM;
  int tid = threadIdx.x;
  float s = 0.f;
  for (int i = tid; i < D_DIM; i += 256) { float v = xr[i]; s += v * v; }
  __shared__ float red[256];
  red[tid] = s; __syncthreads();
  for (int o = 128; o > 0; o >>= 1) { if (tid < o) red[tid] += red[tid + o]; __syncthreads(); }
  float inv = rsqrtf(red[0] / (float)D_DIM + 1e-5f);
  for (int i = tid; i < D_DIM; i += 256)
    out[(size_t)row * D_DIM + i] = (__bf16)(w[i] * xr[i] * inv);
}

// ------- Weight convert+transpose: fp32 in[K,N] -> bf16 out[N,K] -------
__global__ void wconv_kernel(const float* __restrict__ in, __bf16* __restrict__ out, int K, int N) {
  __shared__ float t[32][33];
  int n0 = blockIdx.x * 32, k0 = blockIdx.y * 32;
  int tx = threadIdx.x, ty = threadIdx.y;
  for (int i = 0; i < 4; ++i)
    t[ty + i * 8][tx] = in[(size_t)(k0 + ty + i * 8) * N + n0 + tx];
  __syncthreads();
  for (int i = 0; i < 4; ++i)
    out[(size_t)(n0 + ty + i * 8) * K + k0 + tx] = (__bf16)t[tx][ty + i * 8];
}

// ---------------- GEMM: C[M,N] = A[M,K](bf16) * Bt[N,K](bf16) ----------------
// Async double-buffered LDS pipeline: stage tile k+1 via global_load_async_to_lds
// while WMMAs consume tile k.  FUSE=1: add fp32 residual R.  OUTBF=1: bf16 output.
template<int FUSE, int OUTBF>
__global__ __launch_bounds__(256) void gemm_kernel(
    const __bf16* __restrict__ A, const __bf16* __restrict__ Bt,
    const float* __restrict__ R, float* __restrict__ outF, __bf16* __restrict__ outB,
    int M, int N, int K) {
  __shared__ __align__(16) __bf16 As[2][128 * 40];
  __shared__ __align__(16) __bf16 Bs[2][128 * 40];
  const int tid = threadIdx.x, lane = tid & 31, wid = tid >> 5;
  const int wm = wid >> 1, wn = wid & 1;          // waves 4x2 -> 32x64 each
  const int r4 = lane & 15, hl = lane >> 4;
  const int m0 = blockIdx.y * 128, n0 = blockIdx.x * 128;

  const int frow = tid >> 2, fc8 = (tid & 3) * 8;            // fill slot 0 (rows 0..63)
  const int frow2 = frow + 64;                               // fill slot 1 (rows 64..127)

  auto issue = [&](int k0, int buf) {                        // 4 async b128 per wave
    async_g2l_b128(As[buf] + frow  * 40 + fc8, A  + (size_t)(m0 + frow ) * K + k0 + fc8);
    async_g2l_b128(Bs[buf] + frow  * 40 + fc8, Bt + (size_t)(n0 + frow ) * K + k0 + fc8);
    async_g2l_b128(As[buf] + frow2 * 40 + fc8, A  + (size_t)(m0 + frow2) * K + k0 + fc8);
    async_g2l_b128(Bs[buf] + frow2 * 40 + fc8, Bt + (size_t)(n0 + frow2) * K + k0 + fc8);
  };

  v8f zero = {};
  v8f acc[2][4];
  for (int i = 0; i < 2; ++i)
    for (int j = 0; j < 4; ++j) acc[i][j] = zero;

  issue(0, 0);
  const int nk = K / 32;
  for (int kt = 0; kt < nk; ++kt) {
    int cur = kt & 1;
    if (kt + 1 < nk) {
      issue((kt + 1) * 32, cur ^ 1);
      WAIT_ASYNCCNT(4);                 // tile kt resident (in-order completion)
    } else {
      WAIT_ASYNCCNT(0);
    }
    __syncthreads();
    v16bf bfr[4];
    for (int j = 0; j < 4; ++j) bfr[j] = load_frag_b(Bs[cur], wn * 64 + j * 16, 0, 40, lane);
    for (int i = 0; i < 2; ++i) {
      v16bf af = load_frag_a(As[cur], wm * 32 + i * 16, 0, 40, lane);
      for (int j = 0; j < 4; ++j) acc[i][j] = wmma_bf16(af, bfr[j], acc[i][j]);
    }
    __syncthreads();                    // buffer cur free for tile kt+2
  }

  for (int i = 0; i < 2; ++i)
    for (int j = 0; j < 4; ++j) {
      int col = n0 + wn * 64 + j * 16 + r4;
      for (int e = 0; e < 8; ++e) {
        int row = m0 + wm * 32 + i * 16 + hl * 8 + e;
        float v = acc[i][j][e];
        if (FUSE) v += R[(size_t)row * N + col];
        if (OUTBF) outB[(size_t)row * N + col] = (__bf16)v;
        else       outF[(size_t)row * N + col] = v;
      }
    }
}

// ------- RoPE + head split: qkv fp32 [S,3D] -> qr,kr,vb bf16 [H,S,HD] -------
__global__ __launch_bounds__(256) void rope_kernel(
    const float* __restrict__ qkv, const float* __restrict__ fcos, const float* __restrict__ fsin,
    __bf16* __restrict__ qr, __bf16* __restrict__ kr, __bf16* __restrict__ vb) {
  int t = blockIdx.x * 256 + threadIdx.x;          // over S*D
  int s = t / D_DIM, hd = t % D_DIM;
  int h = hd / HD_DIM, d = hd % HD_DIM;
  float c  = fcos[(size_t)s * HD_DIM + d];
  float sn = fsin[(size_t)s * HD_DIM + d];
  const float* row = qkv + (size_t)s * (3 * D_DIM);
  float qv = row[hd];
  float qo = (d < 64) ? -row[hd + 64] : row[hd - 64];
  float kv = row[D_DIM + hd];
  float ko = (d < 64) ? -row[D_DIM + hd + 64] : row[D_DIM + hd - 64];
  size_t oi = ((size_t)h * S_LEN + s) * HD_DIM + d;
  qr[oi] = (__bf16)(qv * c + qo * sn);
  kr[oi] = (__bf16)(kv * c + ko * sn);
  vb[oi] = (__bf16)row[2 * D_DIM + hd];
}

// ---------------- Attention: per (64-query tile, head), online softmax ----------------
__global__ __launch_bounds__(256) void attn_kernel(
    const __bf16* __restrict__ qr, const __bf16* __restrict__ kr,
    const __bf16* __restrict__ vb, __bf16* __restrict__ attn_out) {
  const int qt = blockIdx.x, h = blockIdx.y;
  const int tid = threadIdx.x, lane = tid & 31, wid = tid >> 5;
  const int wm = wid >> 1, wn = wid & 1;           // 4x2 waves
  const int r4 = lane & 15, hl = lane >> 4;

  __shared__ __align__(16) unsigned char smem[54016];
  __bf16* Qs = (__bf16*)smem;                      // [64][136]
  __bf16* Ks = (__bf16*)(smem + 17408);            // [64][136]
  __bf16* Ps = (__bf16*)(smem + 17408);            // [64][72]  (aliases Ks)
  float*  Ss = (float*)(smem + 34816);             // [64][68]
  __bf16* Vt = (__bf16*)(smem + 34816);            // [128][72] (aliases Ss)
  float*  mrow = (float*)(smem + 53248);
  float*  lrow = mrow + 64;
  float*  crow = mrow + 128;

  const int trow = tid >> 2, tc8 = (tid & 3) * 8;  // tile fill: 4 async b128 / wave

  { // async load Q tile (64 x 128 bf16)
    const __bf16* src = qr + ((size_t)(h * S_LEN + qt * 64)) * HD_DIM;
    async_g2l_b128(Qs + trow * 136 + tc8,      src + (size_t)trow * HD_DIM + tc8);
    async_g2l_b128(Qs + trow * 136 + tc8 + 32, src + (size_t)trow * HD_DIM + tc8 + 32);
    async_g2l_b128(Qs + trow * 136 + tc8 + 64, src + (size_t)trow * HD_DIM + tc8 + 64);
    async_g2l_b128(Qs + trow * 136 + tc8 + 96, src + (size_t)trow * HD_DIM + tc8 + 96);
  }
  if (tid < 64) { mrow[tid] = -3.0e38f; lrow[tid] = 0.f; }

  v8f zero = {};
  v8f oacc[4];
  for (int j = 0; j < 4; ++j) oacc[j] = zero;
  const float scale = 0.08838834764831845f;        // 1/sqrt(128)

  for (int kt = 0; kt <= qt; ++kt) {
    __syncthreads();                               // Ps readers done before Ks overwrite
    { // async load K tile
      const __bf16* ksrc = kr + ((size_t)(h * S_LEN + kt * 64)) * HD_DIM;
      async_g2l_b128(Ks + trow * 136 + tc8,      ksrc + (size_t)trow * HD_DIM + tc8);
      async_g2l_b128(Ks + trow * 136 + tc8 + 32, ksrc + (size_t)trow * HD_DIM + tc8 + 32);
      async_g2l_b128(Ks + trow * 136 + tc8 + 64, ksrc + (size_t)trow * HD_DIM + tc8 + 64);
      async_g2l_b128(Ks + trow * 136 + tc8 + 96, ksrc + (size_t)trow * HD_DIM + tc8 + 96);
    }
    WAIT_ASYNCCNT(0);
    __syncthreads();

    // S = Q * K^T  (64x64), each wave does 16x32
    v8f sacc[2]; sacc[0] = zero; sacc[1] = zero;
    for (int ks = 0; ks < 4; ++ks) {
      v16bf af = load_frag_a(Qs, wm * 16, ks * 32, 136, lane);
      for (int t = 0; t < 2; ++t) {
        v16bf bg = load_frag_b(Ks, wn * 32 + t * 16, ks * 32, 136, lane);
        sacc[t] = wmma_bf16(af, bg, sacc[t]);
      }
    }
    for (int t = 0; t < 2; ++t) {
      int lcol = wn * 32 + t * 16 + r4;
      for (int e = 0; e < 8; ++e) {
        int lr = wm * 16 + hl * 8 + e;
        float sv = sacc[t][e] * scale;
        if (kt == qt && lcol > lr) sv = -1.0e9f;   // causal mask (diagonal tile only)
        Ss[lr * 68 + lcol] = sv;
      }
    }
    __syncthreads();

    // online softmax: thread r owns row r
    if (tid < 64) {
      int r = tid;
      float mo = mrow[r], mn = mo;
      for (int j = 0; j < 64; ++j) { float v = Ss[r * 68 + j]; mn = v > mn ? v : mn; }
      float cf = __expf(mo - mn);
      float ssum = 0.f;
      for (int j = 0; j < 64; ++j) {
        float p = __expf(Ss[r * 68 + j] - mn);
        ssum += p;
        Ps[r * 72 + j] = (__bf16)p;
      }
      mrow[r] = mn; lrow[r] = lrow[r] * cf + ssum; crow[r] = cf;
    }
    __syncthreads();

    // load V tile transposed (aliases Ss region, now consumed)
    const __bf16* vsrc = vb + ((size_t)(h * S_LEN + kt * 64)) * HD_DIM;
    for (int i = 0; i < 32; ++i) {
      int e = tid + 256 * i;                       // 0..8191
      int key = e >> 7, d = e & 127;
      Vt[d * 72 + key] = vsrc[(size_t)key * HD_DIM + d];
    }
    __syncthreads();

    // rescale O, then O += P * V  (64x128, K=64), each wave 16x64
    for (int e = 0; e < 8; ++e) {
      float cf = crow[wm * 16 + hl * 8 + e];
      for (int j = 0; j < 4; ++j) oacc[j][e] *= cf;
    }
    for (int ks = 0; ks < 2; ++ks) {
      v16bf pf = load_frag_a(Ps, wm * 16, ks * 32, 72, lane);
      for (int j = 0; j < 4; ++j) {
        v16bf vf = load_frag_b(Vt, wn * 64 + j * 16, ks * 32, 72, lane);
        oacc[j] = wmma_bf16(pf, vf, oacc[j]);
      }
    }
  }
  __syncthreads();

  for (int e = 0; e < 8; ++e) {
    int lr = wm * 16 + hl * 8 + e;
    float inv = 1.0f / lrow[lr];
    int srow = qt * 64 + lr;
    for (int j = 0; j < 4; ++j) {
      int d = wn * 64 + j * 16 + r4;
      attn_out[(size_t)srow * D_DIM + h * HD_DIM + d] = (__bf16)(oacc[j][e] * inv);
    }
  }
}

// ---------------- SwiGLU elementwise: h = silu(g1) * g3 ----------------
__global__ __launch_bounds__(256) void silu_mul_kernel(
    const __bf16* __restrict__ g13, __bf16* __restrict__ hbuf) {
  int t = blockIdx.x * 256 + threadIdx.x;          // over S*I
  int s = t / I_DIM, i = t % I_DIM;
  float a = (float)g13[(size_t)s * (2 * I_DIM) + i];
  float b = (float)g13[(size_t)s * (2 * I_DIM) + I_DIM + i];
  hbuf[t] = (__bf16)((a / (1.f + __expf(-a))) * b);
}

extern "C" void kernel_launch(void* const* d_in, const int* in_sizes, int n_in,
                              void* d_out, int out_size, void* d_ws, size_t ws_size,
                              hipStream_t stream) {
  (void)in_sizes; (void)n_in; (void)out_size; (void)ws_size;
  const float* x    = (const float*)d_in[0];
  // d_in[1] = attention_mask (causal; applied analytically in attn_kernel)
  const float* fcos = (const float*)d_in[2];
  const float* fsin = (const float*)d_in[3];
  const float* wq   = (const float*)d_in[4];
  const float* wk   = (const float*)d_in[5];
  const float* wv   = (const float*)d_in[6];
  const float* wo   = (const float*)d_in[7];
  const float* w1   = (const float*)d_in[8];
  const float* w2   = (const float*)d_in[9];
  const float* w3   = (const float*)d_in[10];
  const float* anw  = (const float*)d_in[11];
  const float* fnw  = (const float*)d_in[12];

  size_t off = 0;
  char* base = (char*)d_ws;
  auto carve = [&](size_t bytes) -> char* {
    char* p = base + off;
    off = (off + bytes + 255) & ~(size_t)255;
    return p;
  };
  const size_t D = D_DIM, S = S_LEN, I = I_DIM;
  __bf16* WqkvT = (__bf16*)carve(3 * D * D * 2);   // [6144, 2048]
  __bf16* WoT   = (__bf16*)carve(D * D * 2);       // [2048, 2048]
  __bf16* W13T  = (__bf16*)carve(2 * D * I * 2);   // [16384, 2048]
  __bf16* W2T   = (__bf16*)carve(I * D * 2);       // [2048, 8192]
  __bf16* nx    = (__bf16*)carve(S * D * 2);
  float*  qkv   = (float*) carve(S * 3 * D * 4);
  __bf16* qrb   = (__bf16*)carve(S * D * 2);
  __bf16* krb   = (__bf16*)carve(S * D * 2);
  __bf16* vbb   = (__bf16*)carve(S * D * 2);
  __bf16* attn  = (__bf16*)carve(S * D * 2);
  float*  x2    = (float*) carve(S * D * 4);
  __bf16* g13   = (__bf16*)carve(S * 2 * I * 2);
  __bf16* hb    = (__bf16*)carve(S * I * 2);

  dim3 tb(32, 8);
  // Weight transpose+convert to bf16 [N,K]
  wconv_kernel<<<dim3(64, 64),  tb, 0, stream>>>(wq, WqkvT,               2048, 2048);
  wconv_kernel<<<dim3(64, 64),  tb, 0, stream>>>(wk, WqkvT + D * D,       2048, 2048);
  wconv_kernel<<<dim3(64, 64),  tb, 0, stream>>>(wv, WqkvT + 2 * D * D,   2048, 2048);
  wconv_kernel<<<dim3(64, 64),  tb, 0, stream>>>(wo, WoT,                 2048, 2048);
  wconv_kernel<<<dim3(256, 64), tb, 0, stream>>>(w1, W13T,                2048, 8192);
  wconv_kernel<<<dim3(256, 64), tb, 0, stream>>>(w3, W13T + I * D,        2048, 8192);
  wconv_kernel<<<dim3(64, 256), tb, 0, stream>>>(w2, W2T,                 8192, 2048);

  // Attention half
  rmsnorm_kernel<<<2048, 256, 0, stream>>>(x, anw, nx);
  gemm_kernel<0, 0><<<dim3(48, 16), 256, 0, stream>>>(nx, WqkvT, nullptr, qkv, nullptr,
                                                      2048, 6144, 2048);
  rope_kernel<<<(2048 * 2048) / 256, 256, 0, stream>>>(qkv, fcos, fsin, qrb, krb, vbb);
  attn_kernel<<<dim3(32, 16), 256, 0, stream>>>(qrb, krb, vbb, attn);
  gemm_kernel<1, 0><<<dim3(16, 16), 256, 0, stream>>>(attn, WoT, x, x2, nullptr,
                                                      2048, 2048, 2048);
  // FFN half
  rmsnorm_kernel<<<2048, 256, 0, stream>>>(x2, fnw, nx);
  gemm_kernel<0, 1><<<dim3(128, 16), 256, 0, stream>>>(nx, W13T, nullptr, nullptr, g13,
                                                       2048, 16384, 2048);
  silu_mul_kernel<<<(2048 * 8192) / 256, 256, 0, stream>>>(g13, hb);
  gemm_kernel<1, 0><<<dim3(16, 16), 256, 0, stream>>>(hb, W2T, x2, (float*)d_out, nullptr,
                                                      2048, 2048, 8192);
}
